// SA_60773787239183
// MI455X (gfx1250) — matrix operand order repaired and checked by reading, not compiled, not run on hardware
//
#include <hip/hip_runtime.h>
#include <hip/hip_bf16.h>

typedef __attribute__((ext_vector_type(16))) __bf16 v16bf;
typedef __attribute__((ext_vector_type(8)))  float  v8f;
typedef unsigned short u16;
typedef unsigned int   u32;

#if defined(__has_builtin)
# if __has_builtin(__builtin_amdgcn_global_load_async_to_lds_b128) && \
     __has_builtin(__builtin_amdgcn_s_wait_asynccnt)
#  define USE_ASYNC 1
# endif
#endif
#ifndef USE_ASYNC
# define USE_ASYNC 0
#endif

typedef __attribute__((__vector_size__(16))) int v4i;
#if USE_ASYNC
typedef __attribute__((address_space(1))) v4i gas_v4i;   // global
typedef __attribute__((address_space(3))) v4i las_v4i;   // LDS
#endif

__device__ __forceinline__ u16 f2bf(float f) {
    u32 u = __float_as_uint(f);
    u32 r = u + 0x7fffu + ((u >> 16) & 1u);   // round-to-nearest-even
    return (u16)(r >> 16);
}
__device__ __forceinline__ float bf2f(u16 h) {
    return __uint_as_float(((u32)h) << 16);
}

__device__ __forceinline__ void wait_async_zero() {
#if USE_ASYNC
    __builtin_amdgcn_s_wait_asynccnt(0);
#endif
}

union Frag16 { v16bf v; uint4 q[2]; };

constexpr int GBM = 128, GBN = 64, GBK = 32;
constexpr int GLDA = 40, GLDB = 40;            // padded strides (80 B): conflict-free frag reads

// Stage A tile (128x32 bf16) into LDS buffer. Each thread moves 2x16B.
// Async global->LDS where available; zero-fill for OOB rows / K remainder.
__device__ __forceinline__ void stage_tileA(const u16* __restrict__ A, u16* as_buf,
                                            int blockM, int k0, int M, int K, int tid) {
    int row  = tid >> 1;
    int koff = (tid & 1) * 16;
    int grow = blockM + row;
    u16* dst = &as_buf[row * GLDA + koff];
    bool valid = (grow < M) && ((k0 + koff) < K);   // K is a multiple of 16 everywhere
#if USE_ASYNC
    if (valid) {
        const u16* src = A + (size_t)grow * K + k0 + koff;
        __builtin_amdgcn_global_load_async_to_lds_b128(
            (gas_v4i*)(size_t)src,       (las_v4i*)(v4i*)dst,       0, 0);
        __builtin_amdgcn_global_load_async_to_lds_b128(
            (gas_v4i*)(size_t)(src + 8), (las_v4i*)(v4i*)(dst + 8), 0, 0);
    } else {
        uint4 z = {0,0,0,0};
        reinterpret_cast<uint4*>(dst)[0] = z;
        reinterpret_cast<uint4*>(dst)[1] = z;
    }
#else
    uint4 d0 = {0,0,0,0}, d1 = {0,0,0,0};
    if (valid) {
        const uint4* g = reinterpret_cast<const uint4*>(A + (size_t)grow * K + k0 + koff);
        d0 = g[0]; d1 = g[1];
    }
    uint4* s = reinterpret_cast<uint4*>(dst);
    s[0] = d0; s[1] = d1;
#endif
}

// Stage B tile (32x64 bf16) transposed into LDS: Bs[col][k].
__device__ __forceinline__ void stage_tileB(const u16* __restrict__ B, u16* bs_buf,
                                            int blockN, int k0, int N, int K, int tid) {
    int k  = tid & 31;
    int c0 = (tid >> 5) * 8;
    u16 tmp[8];
    if ((k0 + k) < K) {
        const uint4* g = reinterpret_cast<const uint4*>(B + (size_t)(k0 + k) * N + blockN + c0);
        uint4 d = g[0];
        const u16* u = reinterpret_cast<const u16*>(&d);
        #pragma unroll
        for (int i = 0; i < 8; ++i) tmp[i] = u[i];
    } else {
        #pragma unroll
        for (int i = 0; i < 8; ++i) tmp[i] = 0;
    }
    #pragma unroll
    for (int i = 0; i < 8; ++i) bs_buf[(c0 + i) * GLDB + k] = tmp[i];
}

// ---------------------------------------------------------------------------
// Tiled bf16 WMMA GEMM:  C[M,N] = act(A[M,K] @ B[K,N] + bias)
// Double-buffered LDS, async global->LDS staging for A.
// blockIdx.x -> N tile (fastest varying: co-resident blocks share the A tile
// in L2), blockIdx.y -> M tile.
// 8 waves, each computes a 32x32 patch as 2x2 v_wmma_f32_16x16x32_bf16 tiles.
// ACT: 0=none, 1=relu, 2=elu.  OUT_BF16: store bf16 else f32.
// ---------------------------------------------------------------------------
template<int ACT, bool HAS_BIAS, bool OUT_BF16>
__global__ __launch_bounds__(256)
void gemm_bf16_wmma(const u16* __restrict__ A, const u16* __restrict__ B,
                    const float* __restrict__ bias, void* __restrict__ Cout,
                    int M, int N, int K, int ldc)
{
    __shared__ __align__(16) u16 As[2][GBM * GLDA];
    __shared__ __align__(16) u16 Bs[2][GBN * GLDB];

    const int tid  = threadIdx.x;
    const int lane = tid & 31;
    const int wid  = tid >> 5;
    const int m0   = (wid >> 1) * 32;
    const int n0   = (wid & 1) * 32;
    const int blockN = blockIdx.x * GBN;
    const int blockM = blockIdx.y * GBM;

    v8f acc[2][2] = {};

    const int nchunks = (K + GBK - 1) / GBK;
    stage_tileA(A, As[0], blockM, 0, M, K, tid);
    stage_tileB(B, Bs[0], blockN, 0, N, K, tid);
    wait_async_zero();
    __syncthreads();

    int cur = 0;
    for (int kc = 0; kc < nchunks; ++kc) {
        const bool more = (kc + 1) < nchunks;
        if (more) {   // prefetch next chunk into the other buffer (async path)
            stage_tileA(A, As[cur ^ 1], blockM, (kc + 1) * GBK, M, K, tid);
            stage_tileB(B, Bs[cur ^ 1], blockN, (kc + 1) * GBK, N, K, tid);
        }

        // ---- load fragments per ISA 7.12.2 layouts, compute 4 WMMAs
        Frag16 a[2], b[2];
        {
            const int r  = lane & 15;
            const int hi = lane >> 4;
            const int ka = hi * 8;     // A: lanes0-15 K0..7/16..23, lanes16-31 K8..15/24..31
            #pragma unroll
            for (int mi = 0; mi < 2; ++mi) {
                const u16* base = &As[cur][(m0 + mi * 16 + r) * GLDA];
                a[mi].q[0] = *reinterpret_cast<const uint4*>(base + ka);
                a[mi].q[1] = *reinterpret_cast<const uint4*>(base + 16 + ka);
            }
            const int kb = hi * 8;     // B: lanes0-15 K0..7/16..23, lanes16-31 K8..15/24..31 (per column)
            #pragma unroll
            for (int ni = 0; ni < 2; ++ni) {
                const uint4* p = reinterpret_cast<const uint4*>(&Bs[cur][(n0 + ni * 16 + r) * GLDB + kb]);
                b[ni].q[0] = p[0];
                b[ni].q[1] = p[2];
            }
        }
        #pragma unroll
        for (int mi = 0; mi < 2; ++mi)
            #pragma unroll
            for (int ni = 0; ni < 2; ++ni)
                acc[mi][ni] = __builtin_amdgcn_wmma_f32_16x16x32_bf16(
                    false, a[mi].v, false, b[ni].v, (short)0, acc[mi][ni], false, false);

        if (more) {
            wait_async_zero();         // next buffer fully landed in LDS
            __syncthreads();           // everyone done reading cur + staging done
            cur ^= 1;
        }
    }

    // ---- epilogue: C/D layout -> row = r + 8*(lane>>4), col = lane&15
    const int r      = lane & 15;
    const int rowAdd = (lane >> 4) * 8;
    #pragma unroll
    for (int mi = 0; mi < 2; ++mi) {
        #pragma unroll
        for (int ni = 0; ni < 2; ++ni) {
            int col = blockN + n0 + ni * 16 + r;
            float bv = HAS_BIAS ? bias[col] : 0.0f;
            #pragma unroll
            for (int rr = 0; rr < 8; ++rr) {
                int row = blockM + m0 + mi * 16 + rowAdd + rr;
                if (row < M) {
                    float v = acc[mi][ni][rr] + bv;
                    if (ACT == 1)      v = v > 0.f ? v : 0.f;
                    else if (ACT == 2) v = v > 0.f ? v : (__expf(v) - 1.f);
                    if (OUT_BF16)
                        reinterpret_cast<u16*>(Cout)[(size_t)row * ldc + col] = f2bf(v);
                    else
                        reinterpret_cast<float*>(Cout)[(size_t)row * ldc + col] = v;
                }
            }
        }
    }
}

// ---------------------------------------------------------------------------
// small helper kernels
// ---------------------------------------------------------------------------
__global__ void cvt_f32_bf16(const float* __restrict__ in, u16* __restrict__ out, size_t n) {
    size_t i = (size_t)blockIdx.x * blockDim.x + threadIdx.x;
    if (i < n) out[i] = f2bf(in[i]);
}

// out[k, 0:H] = wl[k, :], out[k, H:2H] = wr[k, :]   (bf16)
__global__ void wcat_bf16(const float* __restrict__ wl, const float* __restrict__ wr,
                          u16* __restrict__ out, int K, int H) {
    size_t i = (size_t)blockIdx.x * blockDim.x + threadIdx.x;
    size_t total = (size_t)K * 2 * H;
    if (i >= total) return;
    int k = (int)(i / (2 * H)), j = (int)(i % (2 * H));
    float v = (j < H) ? wl[(size_t)k * H + j] : wr[(size_t)k * H + (j - H)];
    out[i] = f2bf(v);
}

__global__ void deg_count(const int* __restrict__ dst, float* __restrict__ deg, int n) {
    int i = blockIdx.x * blockDim.x + threadIdx.x;
    if (i < n) atomicAdd(&deg[dst[i]], 1.0f);
}

// agg[dst, :F] += h[src, :F]  (h bf16, row stride ldh), one block per edge
__global__ void edge_scatter(const u16* __restrict__ h, const int* __restrict__ src,
                             const int* __restrict__ dst, float* __restrict__ agg,
                             int F, int ldh) {
    int e = blockIdx.x;
    int j = threadIdx.x * 4;
    if (j >= F) return;
    int s = src[e], d = dst[e];
    uint2 v = *reinterpret_cast<const uint2*>(h + (size_t)s * ldh + j);
    float* a = agg + (size_t)d * F + j;
    atomicAdd(a + 0, bf2f((u16)(v.x & 0xffff)));
    atomicAdd(a + 1, bf2f((u16)(v.x >> 16)));
    atomicAdd(a + 2, bf2f((u16)(v.y & 0xffff)));
    atomicAdd(a + 3, bf2f((u16)(v.y >> 16)));
}

// out = relu(agg/max(deg,1) + bias + root),  root = hr[row, hroff + j] (bf16)
template<bool OUT_BF16>
__global__ void sage_combine(const float* __restrict__ agg, const float* __restrict__ deg,
                             const float* __restrict__ bias, const u16* __restrict__ hr,
                             int ldh, int hroff, void* __restrict__ outp, int Nn, int F) {
    size_t i = (size_t)blockIdx.x * blockDim.x + threadIdx.x;
    if (i >= (size_t)Nn * F) return;
    int row = (int)(i / F), j = (int)(i % F);
    float dv = deg[row]; dv = dv > 1.f ? dv : 1.f;
    float v = agg[i] / dv + bias[j] + bf2f(hr[(size_t)row * ldh + hroff + j]);
    v = v > 0.f ? v : 0.f;
    if (OUT_BF16) reinterpret_cast<u16*>(outp)[i] = f2bf(v);
    else          reinterpret_cast<float*>(outp)[i] = v;
}

// gather rows of f32 matrix by index -> bf16, row length D
__global__ void gather_rows_bf16(const float* __restrict__ x, const int* __restrict__ idx,
                                 u16* __restrict__ out, int D, int n) {
    size_t i = (size_t)blockIdx.x * blockDim.x + threadIdx.x;
    if (i >= (size_t)n * D) return;
    int r = (int)(i / D), c = (int)(i % D);
    out[i] = f2bf(x[(size_t)idx[r] * D + c]);
}

// xcat[r, coff + c] = bf16(src[idx[r], c]), c < Fsrc, xcat row stride ldx
__global__ void gather_concat_bf16(const float* __restrict__ srcm, const int* __restrict__ idx,
                                   u16* __restrict__ xcat, int Fsrc, int ldx, int coff, int n) {
    size_t i = (size_t)blockIdx.x * blockDim.x + threadIdx.x;
    if (i >= (size_t)n * Fsrc) return;
    int r = (int)(i / Fsrc), c = (int)(i % Fsrc);
    xcat[(size_t)r * ldx + coff + c] = f2bf(srcm[(size_t)idx[r] * Fsrc + c]);
}

// out[row] = sum_t bf2f(y[row,t]) * w[t] + b[0]    (D = 384)
__global__ __launch_bounds__(384)
void matvec_out(const u16* __restrict__ y, const float* __restrict__ w,
                const float* __restrict__ b, float* __restrict__ out, int D) {
    __shared__ float red[512];
    int row = blockIdx.x, t = threadIdx.x;
    red[t] = bf2f(y[(size_t)row * D + t]) * w[t];
    if (t < 128) red[384 + t] = 0.f;
    __syncthreads();
    for (int s = 256; s > 0; s >>= 1) {
        if (t < s) red[t] += red[t + s];
        __syncthreads();
    }
    if (t == 0) out[row] = red[0] + b[0];
}

// ---------------------------------------------------------------------------
// host launcher
// ---------------------------------------------------------------------------
static inline size_t alignup(size_t x) { return (x + 255) & ~(size_t)255; }

extern "C" void kernel_launch(void* const* d_in, const int* in_sizes, int n_in,
                              void* d_out, int out_size, void* d_ws, size_t ws_size,
                              hipStream_t stream) {
    const int* drug       = (const int*)d_in[0];
    const int* cell       = (const int*)d_in[1];
    const float* drug_x   = (const float*)d_in[2];
    const float* cell_x   = (const float*)d_in[3];
    const int* edges      = (const int*)d_in[4];
    const float* demb_W   = (const float*)d_in[5];
    const float* demb_b   = (const float*)d_in[6];
    const float* c1_lW    = (const float*)d_in[7];
    const float* c1_lb    = (const float*)d_in[8];
    const float* c1_rW    = (const float*)d_in[9];
    const float* c2_lW    = (const float*)d_in[10];
    const float* c2_lb    = (const float*)d_in[11];
    const float* c2_rW    = (const float*)d_in[12];
    const float* reg_W1   = (const float*)d_in[13];
    const float* reg_b1   = (const float*)d_in[14];
    const float* reg_W2   = (const float*)d_in[15];
    const float* reg_b2   = (const float*)d_in[16];
    const float* reg_W3   = (const float*)d_in[17];
    const float* reg_b3   = (const float*)d_in[18];

    const int D_DRUG = 384, D_CELL = 3504;
    const int B   = in_sizes[0];
    const int NC  = in_sizes[3] / D_CELL;           // 20000
    const int NE  = in_sizes[4] / 2;                // 1280000
    const int* esrc = edges;
    const int* edst = edges + NE;

    // workspace layout
    char* ws = (char*)d_ws;
    size_t off = 0;
    auto take = [&](size_t bytes) { size_t o = off; off = alignup(off + bytes); return o; };

    float* deg     = (float*)(ws + take((size_t)NC * 4));
    u16*   W1cat   = (u16*)  (ws + take((size_t)D_CELL * 2048 * 2));
    u16*   W2cat   = (u16*)  (ws + take((size_t)1024 * 512 * 2));
    u16*   dembWb  = (u16*)  (ws + take((size_t)384 * 128 * 2));
    u16*   rW1b    = (u16*)  (ws + take((size_t)384 * 384 * 2));
    u16*   rW2b    = (u16*)  (ws + take((size_t)384 * 384 * 2));
    u16*   cxb     = (u16*)  (ws + take((size_t)NC * D_CELL * 2));
    u16*   h1      = (u16*)  (ws + take((size_t)NC * 2048 * 2));
    float* agg1    = (float*)(ws + take((size_t)NC * 1024 * 4));
    u16*   c1      = (u16*)  (ws + take((size_t)NC * 1024 * 2));
    u16*   h2      = (u16*)  (ws + take((size_t)NC * 512 * 2));
    float* agg2    = (float*)(ws + take((size_t)NC * 256 * 4));
    float* c2      = (float*)(ws + take((size_t)NC * 256 * 4));
    u16*   drugA   = (u16*)  (ws + take((size_t)B * 384 * 2));
    u16*   xcat    = (u16*)  (ws + take((size_t)B * 384 * 2));
    u16*   y1      = (u16*)  (ws + take((size_t)B * 384 * 2));
    u16*   y2      = (u16*)  (ws + take((size_t)B * 384 * 2));
    (void)ws_size;

    // zero-init accumulators
    (void)hipMemsetAsync(deg,  0, (size_t)NC * 4,        stream);
    (void)hipMemsetAsync(agg1, 0, (size_t)NC * 1024 * 4, stream);
    (void)hipMemsetAsync(agg2, 0, (size_t)NC * 256 * 4,  stream);

    auto g1d = [](size_t n) { return dim3((unsigned)((n + 255) / 256)); };

    // weight / activation conversions
    wcat_bf16<<<g1d((size_t)D_CELL * 2048), 256, 0, stream>>>(c1_lW, c1_rW, W1cat, D_CELL, 1024);
    wcat_bf16<<<g1d((size_t)1024 * 512),   256, 0, stream>>>(c2_lW, c2_rW, W2cat, 1024, 256);
    cvt_f32_bf16<<<g1d((size_t)384 * 128), 256, 0, stream>>>(demb_W, dembWb, (size_t)384 * 128);
    cvt_f32_bf16<<<g1d((size_t)384 * 384), 256, 0, stream>>>(reg_W1, rW1b, (size_t)384 * 384);
    cvt_f32_bf16<<<g1d((size_t)384 * 384), 256, 0, stream>>>(reg_W2, rW2b, (size_t)384 * 384);
    cvt_f32_bf16<<<g1d((size_t)NC * D_CELL), 256, 0, stream>>>(cell_x, cxb, (size_t)NC * D_CELL);
    gather_rows_bf16<<<g1d((size_t)B * 384), 256, 0, stream>>>(drug_x, drug, drugA, D_DRUG, B);

    // in-degree
    deg_count<<<g1d(NE), 256, 0, stream>>>(edst, deg, NE);

    // SAGE layer 1: h1 = cell_x @ [lin_l | lin_r]   (M=NC, K=3504, N=2048)
    {
        dim3 grid(2048 / 64, (NC + 127) / 128);
        gemm_bf16_wmma<0, false, true><<<grid, 256, 0, stream>>>(
            cxb, W1cat, nullptr, h1, NC, 2048, D_CELL, 2048);
    }
    edge_scatter<<<dim3(NE), 256, 0, stream>>>(h1, esrc, edst, agg1, 1024, 2048);
    sage_combine<true><<<g1d((size_t)NC * 1024), 256, 0, stream>>>(
        agg1, deg, c1_lb, h1, 2048, 1024, c1, NC, 1024);

    // SAGE layer 2: h2 = c1 @ [lin_l | lin_r]   (M=NC, K=1024, N=512)
    {
        dim3 grid(512 / 64, (NC + 127) / 128);
        gemm_bf16_wmma<0, false, true><<<grid, 256, 0, stream>>>(
            c1, W2cat, nullptr, h2, NC, 512, 1024, 512);
    }
    edge_scatter<<<dim3(NE), 64, 0, stream>>>(h2, esrc, edst, agg2, 256, 512);
    sage_combine<false><<<g1d((size_t)NC * 256), 256, 0, stream>>>(
        agg2, deg, c2_lb, h2, 512, 256, c2, NC, 256);

    // drug branch into xcat[:, 0:128]: relu(drugA @ demb_W + b)
    {
        dim3 grid(128 / 64, (B + 127) / 128);
        gemm_bf16_wmma<1, true, true><<<grid, 256, 0, stream>>>(
            drugA, dembWb, demb_b, xcat, B, 128, 384, 384);
    }
    // cell branch into xcat[:, 128:384]
    gather_concat_bf16<<<g1d((size_t)B * 256), 256, 0, stream>>>(c2, cell, xcat, 256, 384, 128, B);

    // regression MLP
    {
        dim3 grid(384 / 64, (B + 127) / 128);
        gemm_bf16_wmma<2, true, true><<<grid, 256, 0, stream>>>(
            xcat, rW1b, reg_b1, y1, B, 384, 384, 384);
        gemm_bf16_wmma<2, true, true><<<grid, 256, 0, stream>>>(
            y1, rW2b, reg_b2, y2, B, 384, 384, 384);
    }
    matvec_out<<<dim3(B), 384, 0, stream>>>(y2, reg_W3, reg_b3, (float*)d_out, 384);
}
